// NaiveMHA_7224134992070
// MI455X (gfx1250) — matrix operand order repaired and checked by reading, not compiled, not run on hardware
//
#include <hip/hip_runtime.h>

// MHA for B=2, S=2048, D=1024, H=16, DK=64 on gfx1250 (wave32, WMMA).
// All matmuls use v_wmma_f32_16x16x32_bf16 with f32 accumulation.
// Global->LDS tile staging uses GLOBAL_LOAD_ASYNC_TO_LDS_B128 when the
// toolchain exposes the builtins (guarded; falls back to uint4 copies).

#define TM 128      // rows per block tile (8 waves x 16)
#define KSTEP 32    // contraction step per WMMA

#if defined(__gfx1250__) &&                                            \
    __has_builtin(__builtin_amdgcn_global_load_async_to_lds_b128) &&   \
    __has_builtin(__builtin_amdgcn_s_wait_asynccnt)
#define USE_ASYNC_LDS 1
#else
#define USE_ASYNC_LDS 0
#endif

typedef __attribute__((ext_vector_type(16))) __bf16 v16bf;
typedef __attribute__((ext_vector_type(8)))  float  v8f;
typedef __attribute__((ext_vector_type(4)))  int    v4i;

union FragBF  { v16bf v; __bf16 e[16]; uint4 q[2]; };
union FragHalf { uint4 q; __bf16 e[8]; };

__device__ __forceinline__ v8f v8f_zero() {
  v8f z;
#pragma unroll
  for (int i = 0; i < 8; ++i) z[i] = 0.0f;
  return z;
}

// Copy 16 bytes global -> LDS. Async (ASYNCcnt-tracked DMA) when available.
__device__ __forceinline__ void stage16(const __bf16* g, __bf16* l) {
#if USE_ASYNC_LDS
  __builtin_amdgcn_global_load_async_to_lds_b128(
      (__attribute__((address_space(1))) v4i*)(
          (__attribute__((address_space(1))) void*)g),
      (__attribute__((address_space(3))) v4i*)(
          (__attribute__((address_space(3))) void*)l),
      0, 0);
#else
  *(uint4*)l = *(const uint4*)g;
#endif
}

__device__ __forceinline__ void stage_fence() {
#if USE_ASYNC_LDS
  __builtin_amdgcn_s_wait_asynccnt(0);
#endif
}

__device__ __forceinline__ v8f wmma_bf16(v16bf a, v16bf b, v8f c) {
  // (neg_a, A, neg_b, B, c_mod, C, reuse_a, reuse_b)
  return __builtin_amdgcn_wmma_f32_16x16x32_bf16(false, a, false, b, (short)0, c,
                                                 false, false);
}

// A-fragment (16x32 bf16) from a row-major LDS tile (ld in elements).
// Layout: lane<16 -> K 0..15; lane>=16 -> K 16..31 (contiguous, matches B).
__device__ __forceinline__ v16bf load_afrag(const __bf16* src, int ld, int m,
                                            int kbase, int lane) {
  FragBF f;
  const __bf16* p = src + m * ld + kbase + ((lane < 16) ? 0 : 16);
  f.q[0] = *(const uint4*)(p);
  f.q[1] = *(const uint4*)(p + 8);
  return f.v;
}

// B-fragment (32x16 bf16) from a [n][k] (K-contiguous) LDS tile.
// Layout: n = lane&15, elements e -> K = e + (lane<16 ? 0 : 16).
__device__ __forceinline__ v16bf load_bfrag_kcontig(const __bf16* src, int ld,
                                                    int n, int kbase, int lane) {
  FragBF f;
  const __bf16* p = src + n * ld + kbase + ((lane < 16) ? 0 : 16);
  f.q[0] = *(const uint4*)(p);
  f.q[1] = *(const uint4*)(p + 8);
  return f.v;
}

// B-fragment from a row-major [k][n] LDS tile (strided gather) -- used for V.
__device__ __forceinline__ v16bf load_bfrag_kstride(const __bf16* src, int ld,
                                                    int n, int kbase, int lane) {
  FragBF f;
  const int k0 = kbase + ((lane < 16) ? 0 : 16);
#pragma unroll
  for (int e = 0; e < 16; ++e) f.e[e] = src[(k0 + e) * ld + n];
  return f.v;
}

// ---------------------------------------------------------------- converts
__global__ __launch_bounds__(256) void cvt_f32_bf16(const float* __restrict__ x,
                                                    __bf16* __restrict__ y, int n) {
  int i = blockIdx.x * blockDim.x + threadIdx.x;
  if (i < n) y[i] = (__bf16)x[i];
}

// ------------------------------------------------------- per-head QKV GEMM
// Y[b][h][s][dk] = X[b][s][:] @ W[h][:][dk]   (S x D) @ (D x 64)
__global__ __launch_bounds__(256) void qkv_proj_kernel(
    const __bf16* __restrict__ X,  // [B, S, D]
    const __bf16* __restrict__ W,  // [H, D, 64]
    __bf16* __restrict__ Y,        // [B*H, S, 64]
    int S_, int D_, int H_) {
  const int rb = blockIdx.x;       // 128-row block
  const int bh = blockIdx.y;       // b*H + h
  const int b = bh / H_, h = bh % H_;
  const int tid = threadIdx.x, lane = tid & 31, wave = tid >> 5;

  const __bf16* Ab = X + (size_t)b * S_ * D_ + (size_t)rb * TM * D_;
  const __bf16* Wb = W + (size_t)h * D_ * 64;
  __bf16* Yb = Y + ((size_t)bh * S_ + (size_t)rb * TM) * 64;

  __shared__ __bf16 sA[TM * 40];   // 128 rows x 32 k, ld=40 (pad)
  __shared__ __bf16 sW[64 * 40];   // transposed weights: [n][k], ld=40

  v8f acc[4];
#pragma unroll
  for (int i = 0; i < 4; ++i) acc[i] = v8f_zero();

  for (int k0 = 0; k0 < D_; k0 += KSTEP) {
    __syncthreads();
    {  // stage A tile: async b128 global -> LDS
      int r = tid >> 1, c = (tid & 1) * 16;
      const __bf16* g = Ab + (size_t)r * D_ + k0 + c;
      __bf16* s = sA + r * 40 + c;
      stage16(g, s);
      stage16(g + 8, s + 8);
    }
    {  // stage W tile transposed ([32 k][64 n] -> [n][k])
      int k = tid >> 3, n0 = (tid & 7) * 8;
      FragHalf t;
      t.q = *(const uint4*)(Wb + (size_t)(k0 + k) * 64 + n0);
#pragma unroll
      for (int i = 0; i < 8; ++i) sW[(n0 + i) * 40 + k] = t.e[i];
    }
    stage_fence();
    __syncthreads();
    v16bf af = load_afrag(sA, 40, wave * 16 + (lane & 15), 0, lane);
#pragma unroll
    for (int nt = 0; nt < 4; ++nt) {
      v16bf bf = load_bfrag_kcontig(sW, 40, nt * 16 + (lane & 15), 0, lane);
      acc[nt] = wmma_bf16(af, bf, acc[nt]);
    }
  }
  const int rowoff = wave * 16 + ((lane < 16) ? 0 : 8);
  const int col = lane & 15;
#pragma unroll
  for (int nt = 0; nt < 4; ++nt)
#pragma unroll
    for (int r = 0; r < 8; ++r)
      Yb[(size_t)(rowoff + r) * 64 + nt * 16 + col] = (__bf16)acc[nt][r];
}

// ------------------------------------------------------ flash attention
// One block per (b, h, 128-query block). K/V tiles of 64 rows in LDS.
__global__ __launch_bounds__(256) void flash_attn_kernel(
    const __bf16* __restrict__ Q,   // [B*H, S, 64]
    const __bf16* __restrict__ Km,  // [B*H, S, 64]
    const __bf16* __restrict__ Vm,  // [B*H, S, 64]
    __bf16* __restrict__ O,         // [B, S, H*64] head-concat
    int S_, int H_) {
  const int qb = blockIdx.x;
  const int bh = blockIdx.y;
  const int b = bh / H_, h = bh % H_;
  const int tid = threadIdx.x, lane = tid & 31, wave = tid >> 5;

  const __bf16* Qb = Q + ((size_t)bh * S_ + (size_t)qb * TM) * 64;
  const __bf16* Kb = Km + (size_t)bh * S_ * 64;
  const __bf16* Vb = Vm + (size_t)bh * S_ * 64;

  __shared__ __bf16 sK[64 * 72];      // [s][dk], ld=72 (pad)
  __shared__ __bf16 sV[64 * 72];
  __shared__ __bf16 stage[TM * 72];   // Q staging, then per-wave P slabs

  {  // stage Q block (128 x 64)
    int r = tid >> 1, c = (tid & 1) * 32;
    const __bf16* g = Qb + (size_t)r * 64 + c;
    __bf16* s = stage + r * 72 + c;
    stage16(g, s);
    stage16(g + 8, s + 8);
    stage16(g + 16, s + 16);
    stage16(g + 24, s + 24);
  }
  stage_fence();
  __syncthreads();
  v16bf qf[2];
  qf[0] = load_afrag(stage, 72, wave * 16 + (lane & 15), 0, lane);
  qf[1] = load_afrag(stage, 72, wave * 16 + (lane & 15), 32, lane);

  float mrow[8], lrow[8];
  v8f oacc[4];
#pragma unroll
  for (int r = 0; r < 8; ++r) { mrow[r] = -1.0e30f; lrow[r] = 0.0f; }
#pragma unroll
  for (int nt = 0; nt < 4; ++nt) oacc[nt] = v8f_zero();

  const float scale = 0.125f;  // 1/sqrt(64)

  for (int s0 = 0; s0 < S_; s0 += 64) {
    __syncthreads();  // previous tile's LDS reads done (also orders Q vs P)
    {                 // stage K and V tiles (64 x 64 each) via async b128
      int r = tid >> 1, c = (tid & 1) * 32;
      const __bf16* gk = Kb + (size_t)(s0 + r) * 64 + c;
      __bf16* sk = sK + r * 72 + c;
      stage16(gk, sk);
      stage16(gk + 8, sk + 8);
      stage16(gk + 16, sk + 16);
      stage16(gk + 24, sk + 24);
      const __bf16* gv = Vb + (size_t)(s0 + r) * 64 + c;
      __bf16* sv = sV + r * 72 + c;
      stage16(gv, sv);
      stage16(gv + 8, sv + 8);
      stage16(gv + 16, sv + 16);
      stage16(gv + 24, sv + 24);
    }
    stage_fence();
    __syncthreads();

    // scores: S16x64 = Q(16x64) @ K^T ; K rows are ready-made B fragments
    v8f sc[4];
#pragma unroll
    for (int nt = 0; nt < 4; ++nt) {
      v8f a = v8f_zero();
      a = wmma_bf16(qf[0], load_bfrag_kcontig(sK, 72, nt * 16 + (lane & 15), 0, lane), a);
      a = wmma_bf16(qf[1], load_bfrag_kcontig(sK, 72, nt * 16 + (lane & 15), 32, lane), a);
      sc[nt] = a;
    }

    // online softmax: rows live across 16-lane halves; xor-reduce 1,2,4,8
    float newmax[8];
#pragma unroll
    for (int r = 0; r < 8; ++r) {
      float v0 = fmaxf(fmaxf(sc[0][r], sc[1][r]), fmaxf(sc[2][r], sc[3][r])) * scale;
#pragma unroll
      for (int mk = 1; mk < 16; mk <<= 1) v0 = fmaxf(v0, __shfl_xor(v0, mk, 32));
      newmax[r] = fmaxf(mrow[r], v0);
    }
    float rsum[8];
#pragma unroll
    for (int r = 0; r < 8; ++r) rsum[r] = 0.0f;
#pragma unroll
    for (int nt = 0; nt < 4; ++nt)
#pragma unroll
      for (int r = 0; r < 8; ++r) {
        float p = __expf(sc[nt][r] * scale - newmax[r]);
        sc[nt][r] = p;
        rsum[r] += p;
      }
#pragma unroll
    for (int r = 0; r < 8; ++r) {
#pragma unroll
      for (int mk = 1; mk < 16; mk <<= 1) rsum[r] += __shfl_xor(rsum[r], mk, 32);
      float corr = __expf(mrow[r] - newmax[r]);
      lrow[r] = lrow[r] * corr + rsum[r];
      mrow[r] = newmax[r];
#pragma unroll
      for (int nt = 0; nt < 4; ++nt) oacc[nt][r] *= corr;
    }

    // C-layout P -> bf16 A-layout via per-wave LDS slab (same-wave DS order)
    __bf16* Pst = stage + wave * 1024;  // 16 x 64
    {
      const int ro = (lane < 16) ? 0 : 8, col = lane & 15;
#pragma unroll
      for (int nt = 0; nt < 4; ++nt)
#pragma unroll
        for (int r = 0; r < 8; ++r)
          Pst[(ro + r) * 64 + nt * 16 + col] = (__bf16)sc[nt][r];
    }
    v16bf pf0 = load_afrag(Pst, 64, (lane & 15), 0, lane);
    v16bf pf1 = load_afrag(Pst, 64, (lane & 15), 32, lane);
#pragma unroll
    for (int nt = 0; nt < 4; ++nt) {
      oacc[nt] = wmma_bf16(pf0, load_bfrag_kstride(sV, 72, nt * 16 + (lane & 15), 0, lane), oacc[nt]);
      oacc[nt] = wmma_bf16(pf1, load_bfrag_kstride(sV, 72, nt * 16 + (lane & 15), 32, lane), oacc[nt]);
    }
  }

  // finalize: O[b][q][h*64+dk] = oacc / lrow (head-concat layout)
  {
    const int rowoff = wave * 16 + ((lane < 16) ? 0 : 8);
    const int col = lane & 15;
    __bf16* Ob = O + ((size_t)b * S_ + (size_t)qb * TM) * (64 * H_) + h * 64;
#pragma unroll
    for (int r = 0; r < 8; ++r) {
      float inv = 1.0f / lrow[r];
#pragma unroll
      for (int nt = 0; nt < 4; ++nt)
        Ob[(size_t)(rowoff + r) * (64 * H_) + nt * 16 + col] =
            (__bf16)(oacc[nt][r] * inv);
    }
  }
}

// ----------------------------------------------------- output projection
__global__ __launch_bounds__(256) void out_proj_kernel(
    const __bf16* __restrict__ X,    // [R, D]
    const __bf16* __restrict__ W,    // [D, D]
    const float* __restrict__ bias,  // [D]
    float* __restrict__ Y,           // [R, D]
    int Dm) {
  const int rb = blockIdx.x;
  const int nb = blockIdx.y * 64;
  const int tid = threadIdx.x, lane = tid & 31, wave = tid >> 5;

  __shared__ __bf16 sA[TM * 40];
  __shared__ __bf16 sW[64 * 40];

  const __bf16* Ab = X + (size_t)rb * TM * Dm;

  v8f acc[4];
#pragma unroll
  for (int i = 0; i < 4; ++i) acc[i] = v8f_zero();

  for (int k0 = 0; k0 < Dm; k0 += KSTEP) {
    __syncthreads();
    {
      int r = tid >> 1, c = (tid & 1) * 16;
      const __bf16* g = Ab + (size_t)r * Dm + k0 + c;
      __bf16* s = sA + r * 40 + c;
      stage16(g, s);
      stage16(g + 8, s + 8);
    }
    {
      int k = tid >> 3, n0 = (tid & 7) * 8;
      FragHalf t;
      t.q = *(const uint4*)(W + (size_t)(k0 + k) * Dm + nb + n0);
#pragma unroll
      for (int i = 0; i < 8; ++i) sW[(n0 + i) * 40 + k] = t.e[i];
    }
    stage_fence();
    __syncthreads();
    v16bf af = load_afrag(sA, 40, wave * 16 + (lane & 15), 0, lane);
#pragma unroll
    for (int nt = 0; nt < 4; ++nt) {
      v16bf bf = load_bfrag_kcontig(sW, 40, nt * 16 + (lane & 15), 0, lane);
      acc[nt] = wmma_bf16(af, bf, acc[nt]);
    }
  }
  const int rowoff = wave * 16 + ((lane < 16) ? 0 : 8);
  const int col = lane & 15;
#pragma unroll
  for (int nt = 0; nt < 4; ++nt) {
    float bv = bias[nb + nt * 16 + col];
#pragma unroll
    for (int r = 0; r < 8; ++r)
      Y[((size_t)rb * TM + rowoff + r) * Dm + nb + nt * 16 + col] =
          acc[nt][r] + bv;
  }
}

// ---------------------------------------------------------------- launch
extern "C" void kernel_launch(void* const* d_in, const int* in_sizes, int n_in,
                              void* d_out, int out_size, void* d_ws, size_t ws_size,
                              hipStream_t stream) {
  (void)in_sizes; (void)n_in; (void)out_size; (void)ws_size;
  constexpr int B = 2, S = 2048, D = 1024, H = 16, DK = 64;

  const float* tgt = (const float*)d_in[0];
  const float* mem = (const float*)d_in[1];
  const float* Wq  = (const float*)d_in[2];
  const float* Wk  = (const float*)d_in[3];
  const float* Wv  = (const float*)d_in[4];
  const float* Wo  = (const float*)d_in[5];
  const float* bo  = (const float*)d_in[6];
  float* out = (float*)d_out;

  char* ws = (char*)d_ws;
  size_t off = 0;
  auto take = [&](size_t elems) -> __bf16* {
    __bf16* p = (__bf16*)(ws + off);
    off = (off + elems * sizeof(__bf16) + 255) & ~(size_t)255;
    return p;
  };
  __bf16* tgtb = take((size_t)B * S * D);
  __bf16* memb = take((size_t)B * S * D);
  __bf16* Wqb  = take((size_t)H * D * DK);
  __bf16* Wkb  = take((size_t)H * D * DK);
  __bf16* Wvb  = take((size_t)H * D * DK);
  __bf16* Wob  = take((size_t)D * D);
  __bf16* Qb   = take((size_t)B * H * S * DK);
  __bf16* Kb   = take((size_t)B * H * S * DK);
  __bf16* Vb   = take((size_t)B * H * S * DK);
  __bf16* Ob   = take((size_t)B * S * D);   // total ~56 MB

  auto cvt = [&](const float* s, __bf16* d2, int n) {
    cvt_f32_bf16<<<dim3((n + 255) / 256), dim3(256), 0, stream>>>(s, d2, n);
  };
  cvt(tgt, tgtb, B * S * D);
  cvt(mem, memb, B * S * D);
  cvt(Wq, Wqb, H * D * DK);
  cvt(Wk, Wkb, H * D * DK);
  cvt(Wv, Wvb, H * D * DK);
  cvt(Wo, Wob, D * D);

  dim3 gq(S / TM, B * H);
  qkv_proj_kernel<<<gq, 256, 0, stream>>>(tgtb, Wqb, Qb, S, D, H);
  qkv_proj_kernel<<<gq, 256, 0, stream>>>(memb, Wkb, Kb, S, D, H);
  qkv_proj_kernel<<<gq, 256, 0, stream>>>(memb, Wvb, Vb, S, D, H);

  flash_attn_kernel<<<dim3(S / TM, B * H), 256, 0, stream>>>(Qb, Kb, Vb, Ob, S, H);

  out_proj_kernel<<<dim3((B * S) / TM, D / 64), 256, 0, stream>>>(Ob, Wob, bo, out, D);
}
